// PyramidAttention_68470368633010
// MI455X (gfx1250) — matrix operand order repaired and checked by reading, not compiled, not run on hardware
//
#include <hip/hip_runtime.h>
#include <hip/hip_bf16.h>

// ---------------------------------------------------------------------------
// PyramidAttention, MI455X (gfx1250) WMMA bf16 implementation.
// N=4, C=64, Cr=32, H_l=W_l=96 (P=9216 queries), H_s=W_s=48 (L=2304 keys)
// score GEMM:  [16 x 288] x [288 x 2304]  (v_wmma_f32_16x16x32_bf16)
// value GEMM:  [16 x 2304] x [2304 x 576] (v_wmma_f32_16x16x32_bf16)
// ---------------------------------------------------------------------------

#define NB    4
#define CIN   64
#define CR    32
#define HL    96
#define PQ    (HL*HL)      // 9216 query pixels
#define HS    48
#define LK    (HS*HS)      // 2304 keys
#define DK    288          // 32*9 patch dim (score reduction)
#define DV    576          // 64*9 value dim

typedef __attribute__((ext_vector_type(16))) __bf16 v16bf;
typedef __attribute__((ext_vector_type(8)))  float  v8f;

// A-matrix bf16 16x32 fragment: lane(<16) row m holds K {0..15},
// lane(>=16) holds K {16..31}.  p points at this lane's first K run.
__device__ __forceinline__ v16bf load_frag_a(const __bf16* p) {
    v16bf r;
    ((uint4*)&r)[0] = *(const uint4*)(p);        // K +0..7
    ((uint4*)&r)[1] = *(const uint4*)(p + 8);    // K +8..15
    return r;
}
// B-matrix bf16 32x16 fragment: lane holds 16 consecutive K values of one
// column; p points at the first of 16 contiguous bf16.
__device__ __forceinline__ v16bf load_frag_b(const __bf16* p) {
    v16bf r;
    ((uint4*)&r)[0] = *(const uint4*)(p);
    ((uint4*)&r)[1] = *(const uint4*)(p + 8);
    return r;
}

// ------------------------------ kernel 1: 1x1 conv + PReLU -----------------
__global__ void conv1x1_prelu_kernel(const float* __restrict__ in,
                                     const float* __restrict__ w,
                                     const float* __restrict__ b,
                                     const float* __restrict__ a,
                                     float* __restrict__ out,
                                     int O, int HW) {
    int idx = blockIdx.x * blockDim.x + threadIdx.x;
    int total = NB * O * HW;
    if (idx >= total) return;
    int pix = idx % HW;
    int rem = idx / HW;
    int o = rem % O;
    int n = rem / O;
    const float* ip = in + (size_t)n * CIN * HW + pix;
    const float* wp = w + o * CIN;
    float acc = b[o];
#pragma unroll 8
    for (int c = 0; c < CIN; ++c) acc += ip[(size_t)c * HW] * wp[c];
    float al = a[0];
    out[idx] = acc >= 0.f ? acc : al * acc;
}

// ------------------- kernel 2a: query patches (bf16, Unfold order) ---------
__global__ void build_xpatch_kernel(const float* __restrict__ mb,
                                    __bf16* __restrict__ Xp) {
    long long idx = (long long)blockIdx.x * blockDim.x + threadIdx.x;
    long long total = (long long)NB * PQ * DK;
    if (idx >= total) return;
    int k = (int)(idx % DK);
    long long rem = idx / DK;
    int p = (int)(rem % PQ);
    int n = (int)(rem / PQ);
    int c = k / 9, kk = k % 9;
    int kh = kk / 3 - 1, kw = kk % 3 - 1;
    int h = p / HL + kh, w = p % HL + kw;
    float v = 0.f;
    if (h >= 0 && h < HL && w >= 0 && w < HL)
        v = mb[((size_t)n * CR + c) * PQ + h * HL + w];
    Xp[idx] = (__bf16)v;
}

// --------- kernel 2b: key patches, L2-normalized per patch (bf16) ----------
// one wave32 per key row: lane == channel, 9 taps each (32*9 == 288)
__global__ __launch_bounds__(256) void build_key_kernel(
        const float* __restrict__ ref, __bf16* __restrict__ Kn) {
    int wid = threadIdx.x >> 5, lane = threadIdx.x & 31;
    int row = blockIdx.x * 8 + wid;              // n*LK + l
    if (row >= NB * LK) return;
    int n = row / LK, l = row % LK;
    int ly = l / HS, lx = l % HS;
    const float* rp = ref + ((size_t)n * CR + lane) * LK;
    float v[9];
    float ss = 0.f;
#pragma unroll
    for (int i = 0; i < 9; ++i) {
        int y = ly + i / 3 - 1, x = lx + i % 3 - 1;
        float val = (y >= 0 && y < HS && x >= 0 && x < HS) ? rp[y * HS + x] : 0.f;
        v[i] = val;
        ss += val * val;
    }
#pragma unroll
    for (int off = 16; off > 0; off >>= 1) ss += __shfl_xor(ss, off, 32);
    float inv = 1.0f / (sqrtf(ss) + 1e-4f);
    __bf16* kp = Kn + (size_t)row * DK + lane * 9;
#pragma unroll
    for (int i = 0; i < 9; ++i) kp[i] = (__bf16)(v[i] * inv);
}

// ----- kernel 2c: transposed flipped value matrix VfT [N][576][2304] -------
// VfT[n][o*9+(dh+1)*3+(dw+1)][l] = base[n][o][ly-dh][lx-dw] (0-padded)
__global__ void build_vft_kernel(const float* __restrict__ base,
                                 __bf16* __restrict__ VfT) {
    long long idx = (long long)blockIdx.x * blockDim.x + threadIdx.x;
    long long total = (long long)NB * DV * LK;
    if (idx >= total) return;
    int l = (int)(idx % LK);
    long long rem = idx / LK;
    int v = (int)(rem % DV);
    int n = (int)(rem / DV);
    int o = v / 9, kk = v % 9;
    int dh = kk / 3 - 1, dw = kk % 3 - 1;
    int y = l / HS - dh, x = l % HS - dw;
    float val = 0.f;
    if (y >= 0 && y < HS && x >= 0 && x < HS)
        val = base[((size_t)n * CIN + o) * LK + y * HS + x];
    VfT[idx] = (__bf16)val;
}

// ------------------- kernel 3: fused WMMA attention ------------------------
// grid = N * (PQ/16) workgroups, 128 threads (4 wave32).
// LDS: S fp32 [16][2304] (144KB) + attn bf16 [16][2304] (72KB) + red bufs.
#define SMEM_S_BYTES   (16 * LK * 4)
#define SMEM_A_BYTES   (16 * LK * 2)
#define SMEM_BYTES     (SMEM_S_BYTES + SMEM_A_BYTES + (128 + 16 + 16) * 4)

__global__ __launch_bounds__(128) void attn_kernel(
        const __bf16* __restrict__ Xp,   // [N][PQ][288]
        const __bf16* __restrict__ Kn,   // [N][LK][288]
        const __bf16* __restrict__ VfT,  // [N][576][LK]
        float* __restrict__ U) {         // [N][PQ][576] (already *0.25)
    extern __shared__ char smem[];
    float*  S  = (float*)smem;                       // [16][LK]
    __bf16* AT = (__bf16*)(smem + SMEM_S_BYTES);     // [16][LK]
    float*  red    = (float*)(smem + SMEM_S_BYTES + SMEM_A_BYTES); // [16][8]
    float*  rowmax = red + 128;
    float*  rowsum = rowmax + 16;

    int wg   = blockIdx.x;
    int n    = wg / (PQ / 16);
    int qb   = (wg % (PQ / 16)) * 16;
    int tid  = threadIdx.x;
    int wid  = tid >> 5;
    int lane = tid & 31;
    int m    = lane & 15;      // row (A) / column (B,D)
    int hf   = lane >> 4;      // fragment half

    // ---- query A fragments, 9 k-chunks of 32, kept in VGPRs -------------
    const __bf16* xrow = Xp + ((size_t)n * PQ + qb + m) * DK + hf * 16;
    v16bf Aq[9];
#pragma unroll
    for (int kc = 0; kc < 9; ++kc) Aq[kc] = load_frag_a(xrow + kc * 32);

    // ---- phase 1: scores into LDS (x SOFTMAX_SCALE) ----------------------
    for (int kb = wid; kb < LK / 16; kb += 4) {
        const __bf16* krow = Kn + ((size_t)n * LK + kb * 16 + m) * DK + hf * 16;
        v8f c = {};
#pragma unroll
        for (int kc = 0; kc < 9; ++kc) {
            v16bf B = load_frag_b(krow + kc * 32);
            c = __builtin_amdgcn_wmma_f32_16x16x32_bf16(
                    false, Aq[kc], false, B, (short)0, c, false, false);
        }
#pragma unroll
        for (int r = 0; r < 8; ++r)
            S[(r + hf * 8) * LK + kb * 16 + m] = c[r] * 10.0f;
    }
    __syncthreads();

    // ---- phase 2: softmax over 2304 keys (row = tid&15, 8 segments) ------
    {
        int row = tid & 15, seg = tid >> 4;
        float* srow = S + row * LK + seg * (LK / 8);
        float mx = -3.0e38f;
        for (int i = 0; i < LK / 8; ++i) mx = fmaxf(mx, srow[i]);
        red[row * 8 + seg] = mx;
        __syncthreads();
        if (tid < 16) {
            float v = red[tid * 8];
#pragma unroll
            for (int s = 1; s < 8; ++s) v = fmaxf(v, red[tid * 8 + s]);
            rowmax[tid] = v;
        }
        __syncthreads();
        float rm = rowmax[row];
        float sum = 0.f;
        for (int i = 0; i < LK / 8; ++i) {
            float e = __expf(srow[i] - rm);
            srow[i] = e;
            sum += e;
        }
        red[row * 8 + seg] = sum;
        __syncthreads();
        if (tid < 16) {
            float v = 0.f;
#pragma unroll
            for (int s = 0; s < 8; ++s) v += red[tid * 8 + s];
            rowsum[tid] = v;
        }
        __syncthreads();
        float inv = 1.0f / rowsum[row];
        __bf16* arow = AT + row * LK + seg * (LK / 8);
        for (int i = 0; i < LK / 8; ++i) arow[i] = (__bf16)(srow[i] * inv);
    }
    __syncthreads();

    // ---- phase 3: U = 0.25 * attn @ VfT  (K = 2304) ----------------------
    const __bf16* abase = AT + m * LK + hf * 16;
    for (int nb = wid; nb < DV / 16; nb += 4) {
        const __bf16* vrow = VfT + ((size_t)n * DV + nb * 16 + m) * LK + hf * 16;
        v8f c = {};
        for (int kc = 0; kc < LK / 32; ++kc) {
            v16bf Af = load_frag_a(abase + kc * 32);   // ds_load_b128 x2
            v16bf Bf = load_frag_b(vrow + kc * 32);    // global_load_b128 x2
            c = __builtin_amdgcn_wmma_f32_16x16x32_bf16(
                    false, Af, false, Bf, (short)0, c, false, false);
        }
#pragma unroll
        for (int r = 0; r < 8; ++r)
            U[((size_t)n * PQ + qb + r + hf * 8) * DV + nb * 16 + m] = c[r] * 0.25f;
    }
}

// ---------------- kernel 4: 9-way shifted gather + residual ----------------
__global__ void gather_out_kernel(const float* __restrict__ U,
                                  const float* __restrict__ input_l,
                                  float* __restrict__ out) {
    int idx = blockIdx.x * blockDim.x + threadIdx.x;
    int total = NB * CIN * PQ;
    if (idx >= total) return;
    int p = idx % PQ;
    int rem = idx / PQ;
    int o = rem % CIN;
    int n = rem / CIN;
    int h = p / HL, w = p % HL;
    float acc = 0.f;
#pragma unroll
    for (int dh = -1; dh <= 1; ++dh)
#pragma unroll
        for (int dw = -1; dw <= 1; ++dw) {
            int qh = h + dh, qw = w + dw;
            if (qh >= 0 && qh < HL && qw >= 0 && qw < HL)
                acc += U[((size_t)n * PQ + qh * HL + qw) * DV
                         + o * 9 + (dh + 1) * 3 + (dw + 1)];
        }
    out[idx] = acc + input_l[idx];
}

// ---------------------------------------------------------------------------
extern "C" void kernel_launch(void* const* d_in, const int* in_sizes, int n_in,
                              void* d_out, int out_size, void* d_ws, size_t ws_size,
                              hipStream_t stream) {
    const float* input_l = (const float*)d_in[0];
    const float* input_s = (const float*)d_in[1];
    const float* w_mlb   = (const float*)d_in[2];
    const float* b_mlb   = (const float*)d_in[3];
    const float* a_mlb   = (const float*)d_in[4];
    const float* w_m     = (const float*)d_in[5];
    const float* b_m     = (const float*)d_in[6];
    const float* a_m     = (const float*)d_in[7];
    const float* w_asm   = (const float*)d_in[8];
    const float* b_asm   = (const float*)d_in[9];
    const float* a_asm   = (const float*)d_in[10];
    float* out = (float*)d_out;

    char* ws = (char*)d_ws;
    size_t off = 0;
    auto alloc = [&](size_t bytes) -> void* {
        void* p = ws + off;
        off += (bytes + 255) & ~(size_t)255;
        return p;
    };
    float*  mb   = (float*) alloc((size_t)NB * CR  * PQ * 4);   // match_base
    float*  refb = (float*) alloc((size_t)NB * CR  * LK * 4);   // ref
    float*  base = (float*) alloc((size_t)NB * CIN * LK * 4);   // base
    __bf16* Xp   = (__bf16*)alloc((size_t)NB * PQ  * DK * 2);   // query patches
    __bf16* Kn   = (__bf16*)alloc((size_t)NB * LK  * DK * 2);   // normed keys
    __bf16* VfT  = (__bf16*)alloc((size_t)NB * DV  * LK * 2);   // value^T
    float*  U    = (float*) alloc((size_t)NB * PQ  * DV * 4);   // attn@V

    const int T = 256;
    // 1) three 1x1-conv + PReLU projections
    conv1x1_prelu_kernel<<<(NB * CR  * PQ + T - 1) / T, T, 0, stream>>>(
        input_l, w_mlb, b_mlb, a_mlb, mb,   CR,  PQ);
    conv1x1_prelu_kernel<<<(NB * CR  * LK + T - 1) / T, T, 0, stream>>>(
        input_s, w_m,   b_m,   a_m,   refb, CR,  LK);
    conv1x1_prelu_kernel<<<(NB * CIN * LK + T - 1) / T, T, 0, stream>>>(
        input_s, w_asm, b_asm, a_asm, base, CIN, LK);

    // 2) bf16 operand prep
    {
        long long tot = (long long)NB * PQ * DK;
        build_xpatch_kernel<<<(int)((tot + T - 1) / T), T, 0, stream>>>(mb, Xp);
    }
    build_key_kernel<<<(NB * LK + 7) / 8, 256, 0, stream>>>(refb, Kn);
    {
        long long tot = (long long)NB * DV * LK;
        build_vft_kernel<<<(int)((tot + T - 1) / T), T, 0, stream>>>(base, VfT);
    }

    // 3) fused scores -> softmax -> attn@V (WMMA bf16, 216KB LDS / WG)
    attn_kernel<<<NB * (PQ / 16), 128, SMEM_BYTES, stream>>>(Xp, Kn, VfT, U);

    // 4) shifted gather + residual
    gather_out_kernel<<<(NB * CIN * PQ + T - 1) / T, T, 0, stream>>>(
        U, input_l, out);
}
